// CenterVotingDecoder_65790309040387
// MI455X (gfx1250) — matrix operand order repaired and checked by reading, not compiled, not run on hardware
//
#include <hip/hip_runtime.h>
#include <hip/hip_bf16.h>
#include <math.h>

// ---------------------------------------------------------------------------
// Problem constants (from the reference)
// ---------------------------------------------------------------------------
constexpr int Bn  = 8;
constexpr int Hh  = 320;
constexpr int Ww  = 320;
constexpr int HW  = Hh * Ww;
constexpr int Ll  = 4;
constexpr float LABEL_THRESHOLD = 100.0f;

typedef __attribute__((ext_vector_type(16))) _Float16 v16h;
typedef __attribute__((ext_vector_type(8)))  float    v8f;

// ---------------------------------------------------------------------------
// Weight pre-pack: build exact per-lane B-fragment register images.
// K ordering: k = (kh*3+kw)*CIN + cin  (kernel-tap major, cin minor).
// B (KxN, f16, 32x16 per step) lane layout:
//   lanes 0-15 : N = lane,    VGPR v -> K = 2v, 2v+1
//   lanes 16-31: N = lane-16, VGPR v -> K = 16+2v, 17+2v
// Stored as wfrag[(((nt*KT)+kt)*32 + lane)*8 + v]  (dwords, 32B aligned/lane)
// ---------------------------------------------------------------------------
template<int CIN, int COUT>
__global__ void pack_weights_kernel(const float* __restrict__ w,
                                    unsigned* __restrict__ wfrag, int total) {
    constexpr int KT = (CIN * 9) / 32;
    int idx = blockIdx.x * blockDim.x + threadIdx.x;
    if (idx >= total) return;
    int v    = idx & 7;
    int lane = (idx >> 3) & 31;
    int kt   = (idx >> 8) % KT;
    int nt   = (idx >> 8) / KT;

    int n    = lane & 15;
    int cout = nt * 16 + n;
    int k0   = kt * 32 + ((lane & 16) ? 16 : 0) + 2 * v;

    unsigned pk = 0;
#pragma unroll
    for (int e = 0; e < 2; ++e) {
        int k    = k0 + e;
        int cin  = k % CIN;          // power-of-two CIN -> cheap
        int kidx = k / CIN;          // 0..8 : kh*3+kw
        float f  = w[((size_t)cout * CIN + cin) * 9 + kidx]; // OIHW
        _Float16 hf = (_Float16)f;
        unsigned short bits;
        __builtin_memcpy(&bits, &hf, 2);
        pk |= ((unsigned)bits) << (16 * e);
    }
    wfrag[idx] = pk;
}

// ---------------------------------------------------------------------------
// 3x3 conv (pad=1) + bias + ReLU as implicit GEMM via v_wmma_f32_16x16x32_f16.
// Grid: x = H * (W/64) strips, y = batch.  Block: (COUT/16) waves.
// Each wave owns one 16-channel N tile and 4 M-subtiles of 16 pixels
// (64-pixel strip; W % 64 == 0 so strips never cross rows).
// LDS im2col layout: patch[(row*COLS + col) * CINP + cin], cin contiguous,
// CINP = CIN+8 pad -> per-lane A-fragment = two contiguous 8-half runs
// (cin+0..7, cin+8..15) loaded with two ds_load_b128, bank-conflict-free.
// ---------------------------------------------------------------------------
template<int CIN, int COUT, typename Tin>
__global__ void __launch_bounds__(COUT * 2)
conv3x3_relu_wmma(const Tin* __restrict__ in,
                  const unsigned* __restrict__ wfrag,
                  const float* __restrict__ bias,
                  _Float16* __restrict__ out) {
    constexpr int KT   = (CIN * 9) / 32;
    constexpr int CINP = CIN + 8;            // padded cin pitch
    constexpr int MW   = 64;                 // pixels per strip
    constexpr int COLS = MW + 2;             // strip + 1 halo each side
    constexpr int NSUB = MW / 16;            // 4 WMMA subtiles
    __shared__ alignas(16) _Float16 patch[3 * COLS * CINP];

    const int tile = blockIdx.x;
    const int b    = blockIdx.y;
    const int h    = tile / (Ww / MW);
    const int w0   = (tile - h * (Ww / MW)) * MW;
    const int tid  = threadIdx.x;
    constexpr int NTHREADS = COUT * 2;

    // Cooperative im2col patch: 3 rows x COLS cols x CIN, zero-padded edges.
    // Enumeration keeps consecutive threads on consecutive gx (coalesced).
    for (int e = tid; e < CIN * 3 * COLS; e += NTHREADS) {
        int cin = e / (3 * COLS);
        int rem = e - cin * (3 * COLS);
        int r   = rem / COLS;
        int c   = rem - r * COLS;
        int gy  = h + r - 1;
        int gx  = w0 + c - 1;
        float val = 0.0f;
        if (gy >= 0 && gy < Hh && gx >= 0 && gx < Ww)
            val = (float)in[(((size_t)b * CIN + cin) * Hh + gy) * Ww + gx];
        patch[(r * COLS + c) * CINP + cin] = (_Float16)val;
    }
    __syncthreads();

    const int lane    = tid & 31;
    const int nt      = tid >> 5;          // N tile id
    const int m       = lane & 15;         // pixel row within M subtile
    const int laneHiA = (lane & 16) ? 16 : 0;
    const int n       = lane & 15;
    const int cout    = nt * 16 + n;
    const float bv    = bias[cout];

    // Bias-initialized accumulators (bias is per-N, constant over M).
    v8f acc[NSUB];
#pragma unroll
    for (int s = 0; s < NSUB; ++s)
#pragma unroll
        for (int v = 0; v < 8; ++v)
            acc[s][v] = bv;

    const unsigned* wbase = wfrag + ((size_t)nt * KT * 32 + lane) * 8;

    for (int kt = 0; kt < KT; ++kt) {
        // B fragment: pre-packed register image, 8 consecutive dwords/lane.
        union { v16h h; uint4 u[2]; } bf;
        const uint4* wp = (const uint4*)(wbase + (size_t)kt * 256);
        bf.u[0] = wp[0];
        bf.u[1] = wp[1];
        __builtin_prefetch(wbase + (size_t)(kt + 1) * 256, 0, 1);

        // Uniform kernel tap per K step (32 | CIN).
        const int kidx    = (kt * 32) / CIN;     // kh*3+kw
        const int kh      = kidx / 3;
        const int kw      = kidx - kh * 3;
        const int cinbase = (kt * 32) % CIN;
        const _Float16* rbase =
            patch + (kh * COLS + kw + m) * CINP + cinbase + laneHiA;

#pragma unroll
        for (int s = 0; s < NSUB; ++s) {
            const _Float16* pr = rbase + s * 16 * CINP;
            union { v16h h; uint4 u[2]; } af;
            af.u[0] = *(const uint4*)(pr);        // K offsets 0..7   (v0..v3)
            af.u[1] = *(const uint4*)(pr + 8);    // K offsets 8..15  (v4..v7)
            acc[s] = __builtin_amdgcn_wmma_f32_16x16x32_f16(
                         false, af.h, false, bf.h, (short)0, acc[s],
                         false, false);
        }
    }

    // Epilogue: C/D layout -> per lane the 8 acc VGPRs are 8 consecutive
    // pixels (M = mbase+v), channel N = lane&15.  One 128b store per subtile.
    const int mbase = (lane & 16) ? 8 : 0;
    _Float16* obase = out + (((size_t)b * COUT + cout) * Hh + h) * Ww + w0 + mbase;
#pragma unroll
    for (int s = 0; s < NSUB; ++s) {
        union { uint4 u; _Float16 hh[8]; } st;
#pragma unroll
        for (int v = 0; v < 8; ++v)
            st.hh[v] = (_Float16)fmaxf(acc[s][v], 0.0f);
        *(uint4*)(obase + s * 16) = st.u;
    }
}

// ---------------------------------------------------------------------------
// 1x1 conv 32->2 + bias + L2 normalization (dv output, f32 into d_out).
// ---------------------------------------------------------------------------
__global__ void head_norm_kernel(const _Float16* __restrict__ h3,
                                 const float* __restrict__ w4,
                                 const float* __restrict__ b4,
                                 float* __restrict__ dv) {
    int p = blockIdx.x * blockDim.x + threadIdx.x;
    if (p >= Bn * HW) return;
    int b  = p / HW;
    int hw = p - b * HW;
    const _Float16* base = h3 + (size_t)b * 32 * HW + hw;
    float dx = b4[0], dy = b4[1];
#pragma unroll
    for (int c = 0; c < 32; ++c) {
        float v = (float)base[(size_t)c * HW];
        dx += v * w4[c];
        dy += v * w4[32 + c];
    }
    float nrm = sqrtf(dx * dx + dy * dy);
    float inv = 1.0f / fmaxf(nrm, 1e-12f);
    dv[((size_t)b * 2 + 0) * HW + hw] = dx * inv;
    dv[((size_t)b * 2 + 1) * HW + hw] = dy * inv;
}

// ---------------------------------------------------------------------------
// Hough voting
// ---------------------------------------------------------------------------
__global__ void zero_f32_kernel(float* __restrict__ p, int n) {
    int i = blockIdx.x * blockDim.x + threadIdx.x;
    if (i < n) p[i] = 0.0f;
}

__global__ void hough_vote_kernel(const int* __restrict__ label,
                                  const float* __restrict__ dv,
                                  float* __restrict__ hist) {
    int p = blockIdx.x * blockDim.x + threadIdx.x;
    if (p >= Bn * HW) return;
    int b   = p / HW;
    int hw  = p - b * HW;
    int lbl = label[p];
    if (lbl <= 0) return;                       // weight is 0 for label==0
    float dx = dv[((size_t)b * 2 + 0) * HW + hw];
    float dy = dv[((size_t)b * 2 + 1) * HW + hw];
    int x = hw % Ww;
    int y = hw / Ww;
    int vx = (int)fminf(fmaxf((float)x + dx, 0.0f), (float)(Ww - 1));
    int vy = (int)fminf(fmaxf((float)y + dy, 0.0f), (float)(Hh - 1));
    atomicAdd(&hist[(((size_t)b * Ll + lbl) * Hh + vy) * Ww + vx], 1.0f);
}

__global__ void hough_argmax_kernel(const float* __restrict__ hist,
                                    float* __restrict__ centers) {
    const int bl  = blockIdx.x;                 // 0 .. B*L-1
    const int tid = threadIdx.x;
    const float* h = hist + (size_t)bl * HW;

    float best = -1.0f;
    int bestIdx = 0;
    for (int i = tid; i < HW; i += blockDim.x) {
        float v = h[i];
        if (v > best) { best = v; bestIdx = i; } // strict > keeps first max
    }
    __shared__ float sv[256];
    __shared__ int   si[256];
    sv[tid] = best;
    si[tid] = bestIdx;
    __syncthreads();
    for (int s = 128; s > 0; s >>= 1) {
        if (tid < s) {
            bool take = (sv[tid + s] > sv[tid]) ||
                        (sv[tid + s] == sv[tid] && si[tid + s] < si[tid]);
            if (take) { sv[tid] = sv[tid + s]; si[tid] = si[tid + s]; }
        }
        __syncthreads();
    }
    if (tid == 0) {
        bool valid = sv[0] > LABEL_THRESHOLD;
        int arg = si[0];
        centers[bl * 2 + 0] = valid ? (float)(arg % Ww) : -1.0f;
        centers[bl * 2 + 1] = valid ? (float)(arg / Ww) : -1.0f;
    }
}

// ---------------------------------------------------------------------------
// Launch
// Inputs: 0:x 1:label 2:w1 3:b1 4:w2 5:b2 6:w3 7:b3 8:w4 9:b4
// d_out: dv (B*2*H*W f32) then centers (B*L*2 as f32)
// ---------------------------------------------------------------------------
extern "C" void kernel_launch(void* const* d_in, const int* in_sizes, int n_in,
                              void* d_out, int out_size, void* d_ws, size_t ws_size,
                              hipStream_t stream) {
    (void)in_sizes; (void)n_in; (void)out_size; (void)ws_size;
    char* ws = (char*)d_ws;

    size_t off = 0;
    auto take = [&](size_t bytes) {
        size_t o = off;
        off += (bytes + 255) & ~(size_t)255;
        return o;
    };
    const size_t wf1  = take((size_t)4 * 36 * 32 * 8 * sizeof(unsigned));
    const size_t wf2  = take((size_t)4 * 18 * 32 * 8 * sizeof(unsigned));
    const size_t wf3  = take((size_t)2 * 18 * 32 * 8 * sizeof(unsigned));
    const size_t act1 = take((size_t)Bn * 64 * HW * sizeof(_Float16));
    const size_t act2 = take((size_t)Bn * 64 * HW * sizeof(_Float16));
    const size_t hist = take((size_t)Bn * Ll * HW * sizeof(float));
    const size_t act3 = act1;   // act1 is dead after conv2 -> overlay

    // ---- weight packing (tiny) ----
    {
        int t1 = 4 * 36 * 256;
        pack_weights_kernel<128, 64><<<(t1 + 255) / 256, 256, 0, stream>>>(
            (const float*)d_in[2], (unsigned*)(ws + wf1), t1);
        int t2 = 4 * 18 * 256;
        pack_weights_kernel<64, 64><<<(t2 + 255) / 256, 256, 0, stream>>>(
            (const float*)d_in[4], (unsigned*)(ws + wf2), t2);
        int t3 = 2 * 18 * 256;
        pack_weights_kernel<64, 32><<<(t3 + 255) / 256, 256, 0, stream>>>(
            (const float*)d_in[6], (unsigned*)(ws + wf3), t3);
    }

    // ---- conv stack (implicit GEMM, WMMA f16->f32) ----
    dim3 g(Hh * (Ww / 64), Bn);
    conv3x3_relu_wmma<128, 64, float><<<g, 128, 0, stream>>>(
        (const float*)d_in[0], (const unsigned*)(ws + wf1),
        (const float*)d_in[3], (_Float16*)(ws + act1));
    conv3x3_relu_wmma<64, 64, _Float16><<<g, 128, 0, stream>>>(
        (const _Float16*)(ws + act1), (const unsigned*)(ws + wf2),
        (const float*)d_in[5], (_Float16*)(ws + act2));
    conv3x3_relu_wmma<64, 32, _Float16><<<g, 64, 0, stream>>>(
        (const _Float16*)(ws + act2), (const unsigned*)(ws + wf3),
        (const float*)d_in[7], (_Float16*)(ws + act3));

    // ---- head: 1x1 conv + L2 normalize -> dv ----
    float* dv = (float*)d_out;
    head_norm_kernel<<<(Bn * HW + 255) / 256, 256, 0, stream>>>(
        (const _Float16*)(ws + act3), (const float*)d_in[8],
        (const float*)d_in[9], dv);

    // ---- Hough voting ----
    float* histp = (float*)(ws + hist);
    int nh = Bn * Ll * HW;
    zero_f32_kernel<<<(nh + 255) / 256, 256, 0, stream>>>(histp, nh);
    hough_vote_kernel<<<(Bn * HW + 255) / 256, 256, 0, stream>>>(
        (const int*)d_in[1], dv, histp);
    hough_argmax_kernel<<<Bn * Ll, 256, 0, stream>>>(
        histp, dv + (size_t)Bn * 2 * HW);
}
